// ArceeMamba_64965675319789
// MI455X (gfx1250) — matrix operand order repaired and checked by reading, not compiled, not run on hardware
//
#include <hip/hip_runtime.h>
#include <hip/hip_bf16.h>
#include <cstdint>

#define DEV_INLINE __device__ __forceinline__

typedef __attribute__((ext_vector_type(16))) __bf16 v16bf;
typedef __attribute__((ext_vector_type(8)))  __bf16 v8bf;
typedef __attribute__((ext_vector_type(8)))  float  v8f;

namespace {
constexpr int kB  = 2;
constexpr int kL  = 4096;
constexpr int kDM = 1024;   // d_model
constexpr int kDI = 2048;   // d_inner
constexpr int kDS = 16;     // d_state
constexpr int kDC = 4;      // d_conv
constexpr int kDR = 64;     // dt_rank
constexpr int kNBL = kB * kL;           // 8192 rows
constexpr int kDBL = kDR + 2 * kDS;     // 96 cols of dbl
constexpr int kCS  = 64;                // scan chunk size
constexpr int kNC  = kL / kCS;          // 64 chunks
}

// ---------------------------------------------------------------------------
// CDNA5 async Global<->LDS helpers (CDNA5 ISA).
// Per-lane: LDS[vdst_lane] = MEM[vaddr_lane]; tracked by ASYNCcnt.
// ---------------------------------------------------------------------------
DEV_INLINE uint32_t lds_off_of(const void* p) {
  // Flat pointer to LDS: low 32 bits are the LDS byte address (ISA §10.2).
  return (uint32_t)(uintptr_t)p;
}
DEV_INLINE void async_load_b128(uint32_t ldsDst, const void* gsrc) {
  asm volatile("global_load_async_to_lds_b128 %0, %1, off"
               :: "v"(ldsDst), "v"(gsrc) : "memory");
}
DEV_INLINE void async_store_b128(void* gdst, uint32_t ldsSrc) {
  asm volatile("global_store_async_from_lds_b128 %0, %1, off"
               :: "v"(gdst), "v"(ldsSrc) : "memory");
}
DEV_INLINE void wait_async0() {
  asm volatile("s_wait_asynccnt 0x0" ::: "memory");
}

// ---------------------------------------------------------------------------
// Split fp32 -> bf16 hi + bf16 lo (bf16x3 GEMM inputs).
// ---------------------------------------------------------------------------
__global__ void mamba_split_bf16(const float* __restrict__ in,
                                 __bf16* __restrict__ hi,
                                 __bf16* __restrict__ lo, size_t n) {
  for (size_t i = (size_t)blockIdx.x * blockDim.x + threadIdx.x; i < n;
       i += (size_t)gridDim.x * blockDim.x) {
    float f = in[i];
    __bf16 h = (__bf16)f;
    hi[i] = h;
    lo[i] = (__bf16)(f - (float)h);
  }
}

DEV_INLINE float softplus_f(float x) { return (x > 20.f) ? x : log1pf(expf(x)); }
DEV_INLINE float silu_f(float x) { return x / (1.f + expf(-x)); }

// ---------------------------------------------------------------------------
// bf16x3 GEMM: C[M,N] = A[M,K(lda)] * W[N,K]^T, fp32 accumulate via
// v_wmma_f32_16x16x32_bf16. Each wave owns a 32x32 tile (2x2 WMMA frags);
// block = WM x WN waves. EPI: 0 = none, 1 = softplus(acc + bias[n]).
// ---------------------------------------------------------------------------
template <int WM, int WN, int EPI>
__global__ __launch_bounds__(WM * WN * 32)
void mamba_gemm_bf16x3(const __bf16* __restrict__ aHi, const __bf16* __restrict__ aLo,
                       const __bf16* __restrict__ wHi, const __bf16* __restrict__ wLo,
                       const float* __restrict__ bias, float* __restrict__ C,
                       int M, int N, int K, int lda) {
  const int lane = threadIdx.x & 31;
  const int wave = threadIdx.x >> 5;
  const int wm = wave / WN;
  const int wn = wave % WN;
  const int rB = blockIdx.y * (WM * 32) + wm * 32;
  const int cB = blockIdx.x * (WN * 32) + wn * 32;
  const int l16 = lane & 15;
  const int hiHalf = lane >> 4;
  const int kbA = hiHalf * 8;    // A k-base: 0 or 8
  const int kbB = hiHalf * 8;    // B k-base: 0 or 8

  v8f acc[2][2];
#pragma unroll
  for (int mt = 0; mt < 2; ++mt)
#pragma unroll
    for (int nt = 0; nt < 2; ++nt)
#pragma unroll
      for (int r = 0; r < 8; ++r) acc[mt][nt][r] = 0.0f;

  for (int k0 = 0; k0 < K; k0 += 32) {
    v16bf ah[2], al[2], bh[2], bl[2];
#pragma unroll
    for (int mt = 0; mt < 2; ++mt) {
      const size_t rowOff = (size_t)(rB + mt * 16 + l16) * lda + k0 + kbA;
      const __bf16* pH = aHi + rowOff;
      const __bf16* pL = aLo + rowOff;
      ((v8bf*)&ah[mt])[0] = *(const v8bf*)pH;          // K = kb..kb+7
      ((v8bf*)&ah[mt])[1] = *(const v8bf*)(pH + 16);   // K = kb+16..kb+23
      ((v8bf*)&al[mt])[0] = *(const v8bf*)pL;
      ((v8bf*)&al[mt])[1] = *(const v8bf*)(pL + 16);
    }
#pragma unroll
    for (int nt = 0; nt < 2; ++nt) {
      const size_t colOff = (size_t)(cB + nt * 16 + l16) * K + k0 + kbB;
      const __bf16* pH = wHi + colOff;
      const __bf16* pL = wLo + colOff;
      ((v8bf*)&bh[nt])[0] = *(const v8bf*)pH;          // K = kb..kb+7
      ((v8bf*)&bh[nt])[1] = *(const v8bf*)(pH + 16);   // K = kb+16..kb+23
      ((v8bf*)&bl[nt])[0] = *(const v8bf*)pL;
      ((v8bf*)&bl[nt])[1] = *(const v8bf*)(pL + 16);
    }
#pragma unroll
    for (int mt = 0; mt < 2; ++mt)
#pragma unroll
      for (int nt = 0; nt < 2; ++nt) {
        acc[mt][nt] = __builtin_amdgcn_wmma_f32_16x16x32_bf16(
            false, ah[mt], false, bh[nt], (short)0, acc[mt][nt], false, false);
        acc[mt][nt] = __builtin_amdgcn_wmma_f32_16x16x32_bf16(
            false, ah[mt], false, bl[nt], (short)0, acc[mt][nt], false, false);
        acc[mt][nt] = __builtin_amdgcn_wmma_f32_16x16x32_bf16(
            false, al[mt], false, bh[nt], (short)0, acc[mt][nt], false, false);
      }
  }

  // D layout: VGPR r holds M = r + 8*(lane>=16), N = lane%16.
  const int mOff = hiHalf * 8;
#pragma unroll
  for (int mt = 0; mt < 2; ++mt)
#pragma unroll
    for (int nt = 0; nt < 2; ++nt) {
      const int gc = cB + nt * 16 + l16;
#pragma unroll
      for (int r = 0; r < 8; ++r) {
        const int gr = rB + mt * 16 + mOff + r;
        float v = acc[mt][nt][r];
        if constexpr (EPI == 1) v = softplus_f(v + bias[gc]);
        C[(size_t)gr * N + gc] = v;
      }
    }
}

// ---------------------------------------------------------------------------
// Depthwise causal conv (width 4) + SiLU.
// ---------------------------------------------------------------------------
__global__ void mamba_conv_silu(const float* __restrict__ x,
                                const float* __restrict__ cw,
                                const float* __restrict__ cb,
                                float* __restrict__ xc) {
  size_t i = (size_t)blockIdx.x * blockDim.x + threadIdx.x;
  if (i >= (size_t)kNBL * kDI) return;
  const int dch = (int)(i % kDI);
  const size_t bl = i / kDI;
  const int l = (int)(bl % kL);
  const size_t bOff = (bl - (size_t)l) * kDI;   // (b*L)*DI
  float s = cb[dch];
#pragma unroll
  for (int k = 0; k < kDC; ++k) {
    const int ll = l + k - (kDC - 1);
    if (ll >= 0) s += cw[dch * kDC + k] * x[bOff + (size_t)ll * kDI + dch];
  }
  xc[i] = silu_f(s);
}

// ---------------------------------------------------------------------------
// Chunked selective scan (exact linear-recurrence decomposition).
// h_l = a_l*h_{l-1} + u_l,  a = exp(dt*A), u = dt*x*B.
// Pass 1: per chunk, P = prod(a), H = h given h_in = 0.   (16384 blocks)
// Pass 2: prefix over chunks: h_in[c] = P[c-1]*h_in[c-1] + H[c-1].
// Pass 3: per chunk, recompute from exact h_in and emit y (in-place over dt).
// LDS staging uses global_load_async_to_lds / global_store_async_from_lds.
// ---------------------------------------------------------------------------
__global__ __launch_bounds__(256)
void mamba_scan_pass1(const float* __restrict__ dty,  // dt
                      const float* __restrict__ dbl,  // (NBL,96): B at col 64
                      const float* __restrict__ xc,
                      const float* __restrict__ A_log,
                      float* __restrict__ P, float* __restrict__ H) {
  __shared__ float sdt[kCS][16], sx[kCS][16], sB[kCS][16];
  const int c = blockIdx.x % kNC;
  const int rest = blockIdx.x / kNC;
  const int d0 = (rest % (kDI / 16)) * 16;
  const int b = rest / (kDI / 16);
  const int t = threadIdx.x;
  const int ch = t >> 4;          // channel within block (0..15)
  const int n = t & 15;           // state index (0..15)
  const int d = d0 + ch;
  const float A_dn = -expf(A_log[d * kDS + n]);
  const size_t rowBase = (size_t)b * kL + (size_t)c * kCS;

  {  // async stage: each of 256 lanes deposits one 16B slot into LDS
    const int lr = t >> 2;
    const int cc = (t & 3) * 4;
    const size_t g = rowBase + lr;
    async_load_b128(lds_off_of(&sdt[lr][cc]), dty + g * kDI + d0 + cc);
    async_load_b128(lds_off_of(&sx[lr][cc]),  xc  + g * kDI + d0 + cc);
    async_load_b128(lds_off_of(&sB[lr][cc]),  dbl + g * kDBL + kDR + cc);
  }
  wait_async0();
  __syncthreads();

  float h = 0.f, prod = 1.f;
#pragma unroll 4
  for (int l = 0; l < kCS; ++l) {
    const float dtv = sdt[l][ch];
    const float xv  = sx[l][ch];
    const float bv  = sB[l][n];
    const float a   = expf(dtv * A_dn);
    h = a * h + (dtv * xv) * bv;
    prod *= a;
  }
  const size_t tid = ((size_t)b * kDI + d) * kDS + n;
  P[tid * kNC + c] = prod;
  H[tid * kNC + c] = h;
}

__global__ __launch_bounds__(256)
void mamba_scan_pass2(const float* __restrict__ P, const float* __restrict__ H,
                      float* __restrict__ Hin) {
  const size_t tid = (size_t)blockIdx.x * blockDim.x + threadIdx.x;  // < B*DI*DS
  const float* p = P + tid * kNC;
  const float* hh = H + tid * kNC;
  float* o = Hin + tid * kNC;
  float hin = 0.f;
#pragma unroll 4
  for (int c = 0; c < kNC; ++c) {
    o[c] = hin;
    hin = p[c] * hin + hh[c];
  }
}

__global__ __launch_bounds__(256)
void mamba_scan_pass3(float* __restrict__ dty,        // in: dt, out: y
                      const float* __restrict__ dbl,  // B at 64, C at 80
                      const float* __restrict__ xc,
                      const float* __restrict__ A_log,
                      const float* __restrict__ Hin) {
  __shared__ float sdt[kCS][16], sx[kCS][16], sB[kCS][16], sC[kCS][16], sy[kCS][16];
  const int c = blockIdx.x % kNC;
  const int rest = blockIdx.x / kNC;
  const int d0 = (rest % (kDI / 16)) * 16;
  const int b = rest / (kDI / 16);
  const int t = threadIdx.x;
  const int ch = t >> 4;
  const int n = t & 15;
  const int d = d0 + ch;
  const float A_dn = -expf(A_log[d * kDS + n]);
  const size_t rowBase = (size_t)b * kL + (size_t)c * kCS;
  const size_t tid = ((size_t)b * kDI + d) * kDS + n;

  const int lr = t >> 2;
  const int cc = (t & 3) * 4;
  {
    const size_t g = rowBase + lr;
    async_load_b128(lds_off_of(&sdt[lr][cc]), dty + g * kDI + d0 + cc);
    async_load_b128(lds_off_of(&sx[lr][cc]),  xc  + g * kDI + d0 + cc);
    async_load_b128(lds_off_of(&sB[lr][cc]),  dbl + g * kDBL + kDR + cc);
    async_load_b128(lds_off_of(&sC[lr][cc]),  dbl + g * kDBL + kDR + kDS + cc);
  }
  float h = Hin[tid * kNC + c];
  wait_async0();
  __syncthreads();

#pragma unroll 4
  for (int l = 0; l < kCS; ++l) {
    const float dtv = sdt[l][ch];
    const float xv  = sx[l][ch];
    const float bv  = sB[l][n];
    const float cv  = sC[l][n];
    const float dA  = expf(dtv * A_dn);
    h = dA * h + (dtv * xv) * bv;
    float p = h * cv;
    p += __shfl_xor(p, 8, 16);
    p += __shfl_xor(p, 4, 16);
    p += __shfl_xor(p, 2, 16);
    p += __shfl_xor(p, 1, 16);
    if (n == 0) sy[l][ch] = p;
  }
  __syncthreads();

  // async write-back LDS -> global (completion covered by s_endpgm wait-idle)
  async_store_b128(dty + (rowBase + lr) * kDI + d0 + cc, lds_off_of(&sy[lr][cc]));
}

// ---------------------------------------------------------------------------
// out_z = (y + x_c * D) * silu(z), written over z.
// ---------------------------------------------------------------------------
__global__ void mamba_fuse_gate(const float* __restrict__ y,
                                const float* __restrict__ xc,
                                const float* __restrict__ Dv,
                                float* __restrict__ z) {
  size_t i = (size_t)blockIdx.x * blockDim.x + threadIdx.x;
  if (i >= (size_t)kNBL * kDI) return;
  const int dch = (int)(i % kDI);
  const float yy = y[i] + xc[i] * Dv[dch];
  const float zv = z[i];
  z[i] = yy * silu_f(zv);
}

// ---------------------------------------------------------------------------
// Host-side orchestration.
// ---------------------------------------------------------------------------
extern "C" void kernel_launch(void* const* d_in, const int* in_sizes, int n_in,
                              void* d_out, int out_size, void* d_ws, size_t ws_size,
                              hipStream_t stream) {
  (void)in_sizes; (void)n_in; (void)out_size; (void)ws_size;
  const float* hidden    = (const float*)d_in[0];  // (B,L,DM)
  const float* in_proj_w = (const float*)d_in[1];  // (2*DI, DM)
  const float* conv_w    = (const float*)d_in[2];  // (DI, DC)
  const float* conv_b    = (const float*)d_in[3];  // (DI)
  const float* x_proj_w  = (const float*)d_in[4];  // (96, DI)
  const float* dt_proj_w = (const float*)d_in[5];  // (DI, DR)
  const float* dt_proj_b = (const float*)d_in[6];  // (DI)
  const float* A_log     = (const float*)d_in[7];  // (DI, DS)
  const float* Dvec      = (const float*)d_in[8];  // (DI)
  const float* out_proj_w= (const float*)d_in[9];  // (DM, DI)

  // Workspace carve-out (all 256B aligned).
  char* base = (char*)d_ws;
  size_t off = 0;
  auto carve = [&](size_t bytes) {
    char* p = base + off;
    off += (bytes + 255) & ~(size_t)255;
    return p;
  };
  float*  xbuf  = (float*)carve((size_t)kNBL * kDI * sizeof(float));   // x
  float*  zbuf  = (float*)carve((size_t)kNBL * kDI * sizeof(float));   // z -> out_z
  float*  xcbuf = (float*)carve((size_t)kNBL * kDI * sizeof(float));   // x_c
  float*  dtbuf = (float*)carve((size_t)kNBL * kDI * sizeof(float));   // dt -> y
  float*  dblbuf= (float*)carve((size_t)kNBL * kDBL * sizeof(float));  // dbl
  __bf16* aHi   = (__bf16*)carve((size_t)kNBL * kDI * sizeof(__bf16)); // act split
  __bf16* aLo   = (__bf16*)carve((size_t)kNBL * kDI * sizeof(__bf16));
  __bf16* wHi   = (__bf16*)carve((size_t)2 * kDI * kDM * sizeof(__bf16)); // wt split
  __bf16* wLo   = (__bf16*)carve((size_t)2 * kDI * kDM * sizeof(__bf16));
  const size_t scanStates = (size_t)kB * kDI * kDS;                    // 65536
  float*  Pbuf  = (float*)carve(scanStates * kNC * sizeof(float));
  float*  Hbuf  = (float*)carve(scanStates * kNC * sizeof(float));
  float*  HinBuf= (float*)carve(scanStates * kNC * sizeof(float));

  const dim3 splitGrid(4096), blk256(256);
  const size_t nElemDI = (size_t)kNBL * kDI;
  const unsigned ewGrid = (unsigned)((nElemDI + 255) / 256);
  const unsigned scanBlocks = (unsigned)(kB * (kDI / 16) * kNC);       // 16384

  // 1) in_proj: xz = H @ Win^T  (two GEMMs: x rows 0..DI-1, z rows DI..2DI-1)
  mamba_split_bf16<<<splitGrid, blk256, 0, stream>>>(hidden, aHi, aLo, (size_t)kNBL * kDM);
  mamba_split_bf16<<<splitGrid, blk256, 0, stream>>>(in_proj_w, wHi, wLo, (size_t)2 * kDI * kDM);
  mamba_gemm_bf16x3<2, 4, 0><<<dim3(kDI / 128, kNBL / 64), dim3(256), 0, stream>>>(
      aHi, aLo, wHi, wLo, nullptr, xbuf, kNBL, kDI, kDM, kDM);
  mamba_gemm_bf16x3<2, 4, 0><<<dim3(kDI / 128, kNBL / 64), dim3(256), 0, stream>>>(
      aHi, aLo, wHi + (size_t)kDI * kDM, wLo + (size_t)kDI * kDM, nullptr, zbuf,
      kNBL, kDI, kDM, kDM);

  // 2) depthwise conv + SiLU
  mamba_conv_silu<<<dim3(ewGrid), blk256, 0, stream>>>(xbuf, conv_w, conv_b, xcbuf);

  // 3) x_proj: dbl = x_c @ Wx^T   (N = 96)
  mamba_split_bf16<<<splitGrid, blk256, 0, stream>>>(xcbuf, aHi, aLo, nElemDI);
  mamba_split_bf16<<<splitGrid, blk256, 0, stream>>>(x_proj_w, wHi, wLo, (size_t)kDBL * kDI);
  mamba_gemm_bf16x3<4, 3, 0><<<dim3(1, kNBL / 128), dim3(384), 0, stream>>>(
      aHi, aLo, wHi, wLo, nullptr, dblbuf, kNBL, kDBL, kDI, kDI);

  // 4) dt = softplus(dbl[:, :64] @ Wdt^T + b)   (A strided: lda = 96, K = 64)
  mamba_split_bf16<<<splitGrid, blk256, 0, stream>>>(dblbuf, aHi, aLo, (size_t)kNBL * kDBL);
  mamba_split_bf16<<<splitGrid, blk256, 0, stream>>>(dt_proj_w, wHi, wLo, (size_t)kDI * kDR);
  mamba_gemm_bf16x3<2, 4, 1><<<dim3(kDI / 128, kNBL / 64), dim3(256), 0, stream>>>(
      aHi, aLo, wHi, wLo, dt_proj_b, dtbuf, kNBL, kDI, kDR, kDBL);

  // 5) chunked selective scan (y overwrites dt in-place)
  mamba_scan_pass1<<<dim3(scanBlocks), blk256, 0, stream>>>(
      dtbuf, dblbuf, xcbuf, A_log, Pbuf, Hbuf);
  mamba_scan_pass2<<<dim3((unsigned)(scanStates / 256)), blk256, 0, stream>>>(
      Pbuf, Hbuf, HinBuf);
  mamba_scan_pass3<<<dim3(scanBlocks), blk256, 0, stream>>>(
      dtbuf, dblbuf, xcbuf, A_log, HinBuf);

  // 6) gate: out_z = (y + x_c*D) * silu(z)  (overwrites z)
  mamba_fuse_gate<<<dim3(ewGrid), blk256, 0, stream>>>(dtbuf, xcbuf, Dvec, zbuf);

  // 7) out_proj: out = out_z @ Wout^T
  mamba_split_bf16<<<splitGrid, blk256, 0, stream>>>(zbuf, aHi, aLo, nElemDI);
  mamba_split_bf16<<<splitGrid, blk256, 0, stream>>>(out_proj_w, wHi, wLo, (size_t)kDM * kDI);
  mamba_gemm_bf16x3<2, 4, 0><<<dim3(kDM / 128, kNBL / 64), dim3(256), 0, stream>>>(
      aHi, aLo, wHi, wLo, nullptr, (float*)d_out, kNBL, kDM, kDI, kDI);
}
